// CrossOnlyAttention_38920993636896
// MI455X (gfx1250) — matrix operand order repaired and checked by reading, not compiled, not run on hardware
//
#include <hip/hip_runtime.h>
#include <hip/hip_bf16.h>

typedef __bf16 bf16_t;
typedef __attribute__((ext_vector_type(16))) __bf16 v16bf;
typedef __attribute__((ext_vector_type(8)))  __bf16 v8bf;
typedef __attribute__((ext_vector_type(8)))  float  v8f;

#define T1P   234
#define TT    469          // T1+T2+1
#define TPAD  480          // padded key length (15 * 32)
#define NH    16
#define HD    64
#define BATCH 64

static __device__ __forceinline__ v8f wmma_bf16(v16bf a, v16bf b, v8f c) {
  return __builtin_amdgcn_wmma_f32_16x16x32_bf16(false, a, false, b, (short)0, c,
                                                 false, false);
}

static __device__ __forceinline__ void sched_fence() {
#if __has_builtin(__builtin_amdgcn_sched_barrier)
  __builtin_amdgcn_sched_barrier(0);
#endif
}

// additive mask bias, faithful to reference (float mask used as additive bias)
static __device__ __forceinline__ float mask_bias(int qi, int ki) {
  if (qi == 0 || ki == 0) return 1.0f;
  int ra = (qi - 1) >= T1P;
  int rb = (ki - 1) >= T1P;
  return (ra != rb) ? 1.0f : 0.0f;
}

// ---------------------------------------------------------------------------
// One-time weight transpose: W[K][N] fp32 -> Wt[N][K] bf16 (K-contiguous, the
// exact B-fragment layout). Wt lives in the 192MB L2 for the whole GEMM.
// ---------------------------------------------------------------------------
__global__ __launch_bounds__(256) void transpose_w(const float* __restrict__ W,
                                                   bf16_t* __restrict__ Wt,
                                                   int K, int N) {
  __shared__ bf16_t tile[64][68];  // pitch 68 keeps 8B alignment, no conflicts
  const int tid = threadIdx.x;
  const int lx = tid & 15, ly = tid >> 4;
  const int nb = blockIdx.x * 64, kb = blockIdx.y * 64;
#pragma unroll
  for (int p = 0; p < 4; ++p) {
    int r = p * 16 + ly;  // k offset
    int c = lx * 4;       // n offset
    float4 v = *(const float4*)(W + (size_t)(kb + r) * N + nb + c);
    tile[c + 0][r] = (bf16_t)v.x;
    tile[c + 1][r] = (bf16_t)v.y;
    tile[c + 2][r] = (bf16_t)v.z;
    tile[c + 3][r] = (bf16_t)v.w;
  }
  __syncthreads();
#pragma unroll
  for (int p = 0; p < 4; ++p) {
    int n = p * 16 + ly;
    int k4 = lx * 4;
    *(uint2*)(Wt + (size_t)(nb + n) * K + kb + k4) = *(const uint2*)&tile[n][k4];
  }
}

// ---------------------------------------------------------------------------
// Kernel A: qkv = x @ W_attn + b_attn using pre-transposed bf16 Wt[3072][1024].
// Block 128 thr (4 waves), tile 64 rows x 128 cols, k-step 32. No LDS, no
// barriers: A and B fragments are contiguous global loads (B hits L2).
// Scatter into bf16 Q[B,H,T,64], K[B,H,T,64], Vt[B,H,64,480].
// ---------------------------------------------------------------------------
__global__ __launch_bounds__(128) void qkv_gemm(
    const float* __restrict__ x, const bf16_t* __restrict__ Wt,
    const float* __restrict__ bias, bf16_t* __restrict__ Q,
    bf16_t* __restrict__ K, bf16_t* __restrict__ Vt) {
  const int tid = threadIdx.x;
  const int wave = tid >> 5;
  const int lane = tid & 31;
  const int lm = lane & 15;
  const int half = lane >> 4;
  const int col0 = blockIdx.x * 128;
  const int row0 = blockIdx.y * 64;
  const int rowA = row0 + wave * 16 + lm;
  const float* xrow = x + (size_t)rowA * 1024;
  const bf16_t* wbase = Wt + (size_t)(col0 + lm) * 1024 + half * 16;

  v8f acc[8];
#pragma unroll
  for (int cb = 0; cb < 8; ++cb) acc[cb] = (v8f){};

  for (int kt = 0; kt < 32; ++kt) {
    const int k0 = kt * 32;
    // A loads first (highest latency chain: load + cvt)
    const float* ap = xrow + k0;
    float4 a0 = *(const float4*)(ap + half * 16);
    float4 a1 = *(const float4*)(ap + half * 16 + 4);
    float4 a2 = *(const float4*)(ap + half * 16 + 8);
    float4 a3 = *(const float4*)(ap + half * 16 + 12);
    // 8 B-fragments: contiguous 32B each from transposed weights (L2-resident)
    v16bf Bf[8];
#pragma unroll
    for (int cb = 0; cb < 8; ++cb)
      Bf[cb] = *(const v16bf*)(wbase + (size_t)cb * 16 * 1024 + k0);

    v16bf A;
    A[0] = (bf16_t)a0.x; A[1] = (bf16_t)a0.y; A[2] = (bf16_t)a0.z; A[3] = (bf16_t)a0.w;
    A[4] = (bf16_t)a1.x; A[5] = (bf16_t)a1.y; A[6] = (bf16_t)a1.z; A[7] = (bf16_t)a1.w;
    A[8] = (bf16_t)a2.x; A[9] = (bf16_t)a2.y; A[10] = (bf16_t)a2.z; A[11] = (bf16_t)a2.w;
    A[12] = (bf16_t)a3.x; A[13] = (bf16_t)a3.y; A[14] = (bf16_t)a3.z; A[15] = (bf16_t)a3.w;

    sched_fence();  // keep all loads before the WMMA burst
#pragma unroll
    for (int cb = 0; cb < 8; ++cb) acc[cb] = wmma_bf16(A, Bf[cb], acc[cb]);
  }

  // Epilogue: add bias, scatter to Q / K / Vt in bf16.
  const int which = col0 >> 10;  // 0=q 1=k 2=v (128-col tile never straddles)
  const int baseRow = row0 + wave * 16 + 8 * half;  // C-layout row for r=0
  const int b0 = baseRow / TT;
  const int t0 = baseRow % TT;
#pragma unroll
  for (int cb = 0; cb < 8; ++cb) {
    int c = col0 + cb * 16 + lm;  // global column (C-layout n = lane&15)
    int h = (c >> 6) & 15;
    int d = c & 63;
    float bv = bias[c];
#pragma unroll
    for (int r = 0; r < 8; ++r) {
      int t = t0 + r, b = b0;
      if (t >= TT) { t -= TT; b += 1; }
      bf16_t val = (bf16_t)(acc[cb][r] + bv);
      size_t bh = (size_t)(b * NH + h);
      if (which == 0)      Q[(bh * TT + t) * HD + d] = val;
      else if (which == 1) K[(bh * TT + t) * HD + d] = val;
      else                 Vt[(bh * HD + d) * TPAD + t] = val;
    }
  }
}

// ---------------------------------------------------------------------------
// Kernel B: flash attention per (b,h). 256 thr = 8 independent waves,
// each wave owns 16-query tiles qt = wave, wave+8, ... ; 32 keys per step.
// ---------------------------------------------------------------------------
__global__ __launch_bounds__(256) void attn(
    const bf16_t* __restrict__ Q, const bf16_t* __restrict__ K,
    const bf16_t* __restrict__ Vt, bf16_t* __restrict__ Y) {
  __shared__ bf16_t ldsP[8][16 * 32];  // per-wave P tile (16 q x 32 keys)
  const int tid = threadIdx.x;
  const int wave = tid >> 5;
  const int lane = tid & 31;
  const int lm = lane & 15;
  const int half = lane >> 4;
  const int bh = blockIdx.x;
  const int b = bh >> 4, h = bh & 15;
  const bf16_t* Qp = Q + (size_t)bh * TT * HD;
  const bf16_t* Kp = K + (size_t)bh * TT * HD;
  const bf16_t* Vp = Vt + (size_t)bh * HD * TPAD;
  bf16_t* Pw = &ldsP[wave][0];
  const float scale = 0.125f;  // 1/sqrt(64)

  for (int qt = wave; qt < 30; qt += 8) {
    int trow = qt * 16 + lm;
    if (trow > TT - 1) trow = TT - 1;  // clamp (results discarded on store)
    const bf16_t* qp = Qp + (size_t)trow * HD;
    v8bf q00 = *(const v8bf*)(qp + half * 16);
    v8bf q01 = *(const v8bf*)(qp + half * 16 + 8);
    v8bf q10 = *(const v8bf*)(qp + 32 + half * 16);
    v8bf q11 = *(const v8bf*)(qp + 32 + half * 16 + 8);
    v16bf Qa0, Qa1;
#pragma unroll
    for (int i = 0; i < 8; ++i) {
      Qa0[i] = q00[i]; Qa0[8 + i] = q01[i];
      Qa1[i] = q10[i]; Qa1[8 + i] = q11[i];
    }

    v8f acc[4];
#pragma unroll
    for (int cb = 0; cb < 4; ++cb) acc[cb] = (v8f){};
    float mrow[8], lrow[8], sf[8];
#pragma unroll
    for (int r = 0; r < 8; ++r) { mrow[r] = -1e30f; lrow[r] = 0.0f; }

    for (int j = 0; j < 15; ++j) {  // 15 * 32 = 480 padded keys
      int key0 = j * 32 + lm;       // this lane's key for column-block 0
      int key1 = key0 + 16;         // column-block 1
      int ck0 = key0 > TT - 1 ? TT - 1 : key0;
      int ck1 = key1 > TT - 1 ? TT - 1 : key1;
      // K B-fragments: contiguous 16 bf16 at hd offset half*16 (+32 for chunk1)
      v16bf B00 = *(const v16bf*)(Kp + (size_t)ck0 * HD + half * 16);
      v16bf B01 = *(const v16bf*)(Kp + (size_t)ck0 * HD + 32 + half * 16);
      v16bf B10 = *(const v16bf*)(Kp + (size_t)ck1 * HD + half * 16);
      v16bf B11 = *(const v16bf*)(Kp + (size_t)ck1 * HD + 32 + half * 16);
      sched_fence();
      v8f S0 = (v8f){}, S1 = (v8f){};
      S0 = wmma_bf16(Qa0, B00, S0);
      S0 = wmma_bf16(Qa1, B01, S0);
      S1 = wmma_bf16(Qa0, B10, S1);
      S1 = wmma_bf16(Qa1, B11, S1);

      float s0[8], s1[8];
#pragma unroll
      for (int r = 0; r < 8; ++r) {
        int qi = qt * 16 + 8 * half + r;
        float a = S0[r] * scale + mask_bias(qi, key0);
        float c = S1[r] * scale + mask_bias(qi, key1);
        if (key0 > TT - 1) a = -1e30f;
        if (key1 > TT - 1) c = -1e30f;
        s0[r] = a; s1[r] = c;
      }
      // online softmax stats: row max/sum across the 16 lanes of this half
#pragma unroll
      for (int r = 0; r < 8; ++r) {
        float mx = fmaxf(s0[r], s1[r]);
        mx = fmaxf(mx, __shfl_xor(mx, 1, 16));
        mx = fmaxf(mx, __shfl_xor(mx, 2, 16));
        mx = fmaxf(mx, __shfl_xor(mx, 4, 16));
        mx = fmaxf(mx, __shfl_xor(mx, 8, 16));
        float mnew = fmaxf(mrow[r], mx);
        sf[r] = __expf(mrow[r] - mnew);
        mrow[r] = mnew;
        float p0 = __expf(s0[r] - mnew);
        float p1 = __expf(s1[r] - mnew);
        s0[r] = p0; s1[r] = p1;
        float rs = p0 + p1;
        rs += __shfl_xor(rs, 1, 16);
        rs += __shfl_xor(rs, 2, 16);
        rs += __shfl_xor(rs, 4, 16);
        rs += __shfl_xor(rs, 8, 16);
        lrow[r] = lrow[r] * sf[r] + rs;
      }
      // C-layout P -> A-layout via per-wave LDS round-trip (same-wave order)
#pragma unroll
      for (int r = 0; r < 8; ++r) {
        Pw[(8 * half + r) * 32 + lm]      = (bf16_t)s0[r];
        Pw[(8 * half + r) * 32 + 16 + lm] = (bf16_t)s1[r];
      }
      asm volatile("s_wait_dscnt 0" ::: "memory");
      const bf16_t* pp = Pw + lm * 32;
      v8bf p0v = *(const v8bf*)(pp + half * 16);
      v8bf p1v = *(const v8bf*)(pp + half * 16 + 8);
      v16bf Pa;
#pragma unroll
      for (int i = 0; i < 8; ++i) { Pa[i] = p0v[i]; Pa[8 + i] = p1v[i]; }

#pragma unroll
      for (int cb = 0; cb < 4; ++cb) {
#pragma unroll
        for (int r = 0; r < 8; ++r) acc[cb][r] *= sf[r];
      }
      // V B-fragments from transposed V: contiguous along key dim
      v16bf Vb[4];
#pragma unroll
      for (int cb = 0; cb < 4; ++cb)
        Vb[cb] = *(const v16bf*)(Vp + (size_t)(cb * 16 + lm) * TPAD +
                                 j * 32 + half * 16);
      sched_fence();
#pragma unroll
      for (int cb = 0; cb < 4; ++cb)
        acc[cb] = wmma_bf16(Pa, Vb[cb], acc[cb]);
    }
    // normalize + store y[b, t, h*64 + col] (bf16, [B,T,C] for proj GEMM)
#pragma unroll
    for (int r = 0; r < 8; ++r) {
      int t = qt * 16 + 8 * half + r;
      if (t > TT - 1) continue;
      float inv = 1.0f / lrow[r];
#pragma unroll
      for (int cb = 0; cb < 4; ++cb) {
        Y[((size_t)(b * TT + t)) * 1024 + h * 64 + cb * 16 + lm] =
            (bf16_t)(acc[cb][r] * inv);
      }
    }
  }
}

// ---------------------------------------------------------------------------
// Kernel C: out = Y @ W_proj + b_proj (Y bf16, Wt bf16 [1024][1024], out f32).
// Same no-LDS tiling as kernel A.
// ---------------------------------------------------------------------------
__global__ __launch_bounds__(128) void proj_gemm(
    const bf16_t* __restrict__ Y, const bf16_t* __restrict__ Wt,
    const float* __restrict__ bias, float* __restrict__ out) {
  const int tid = threadIdx.x;
  const int wave = tid >> 5;
  const int lane = tid & 31;
  const int lm = lane & 15;
  const int half = lane >> 4;
  const int col0 = blockIdx.x * 128;
  const int row0 = blockIdx.y * 64;
  const int rowA = row0 + wave * 16 + lm;
  const bf16_t* yrow = Y + (size_t)rowA * 1024;
  const bf16_t* wbase = Wt + (size_t)(col0 + lm) * 1024 + half * 16;

  v8f acc[8];
#pragma unroll
  for (int cb = 0; cb < 8; ++cb) acc[cb] = (v8f){};

  for (int kt = 0; kt < 32; ++kt) {
    const int k0 = kt * 32;
    const bf16_t* ap = yrow + k0;
    v8bf alo = *(const v8bf*)(ap + half * 16);
    v8bf ahi = *(const v8bf*)(ap + half * 16 + 8);
    v16bf Bf[8];
#pragma unroll
    for (int cb = 0; cb < 8; ++cb)
      Bf[cb] = *(const v16bf*)(wbase + (size_t)cb * 16 * 1024 + k0);

    v16bf A;
#pragma unroll
    for (int i = 0; i < 8; ++i) { A[i] = alo[i]; A[8 + i] = ahi[i]; }

    sched_fence();
#pragma unroll
    for (int cb = 0; cb < 8; ++cb) acc[cb] = wmma_bf16(A, Bf[cb], acc[cb]);
  }

  const int baseRow = row0 + wave * 16 + 8 * half;
#pragma unroll
  for (int cb = 0; cb < 8; ++cb) {
    float bv = bias[col0 + cb * 16 + lm];
#pragma unroll
    for (int r = 0; r < 8; ++r) {
      out[(size_t)(baseRow + r) * 1024 + col0 + cb * 16 + lm] = acc[cb][r] + bv;
    }
  }
}

// ---------------------------------------------------------------------------
extern "C" void kernel_launch(void* const* d_in, const int* in_sizes, int n_in,
                              void* d_out, int out_size, void* d_ws,
                              size_t ws_size, hipStream_t stream) {
  const float* x      = (const float*)d_in[0];
  const float* W_attn = (const float*)d_in[1];
  const float* b_attn = (const float*)d_in[2];
  const float* W_proj = (const float*)d_in[3];
  const float* b_proj = (const float*)d_in[4];
  float* out = (float*)d_out;

  char* ws = (char*)d_ws;
  size_t qBytes  = (size_t)BATCH * NH * TT * HD * sizeof(bf16_t);    // ~61.5 MB
  size_t vBytes  = (size_t)BATCH * NH * HD * TPAD * sizeof(bf16_t);  // ~62.9 MB
  size_t wtABytes = (size_t)3072 * 1024 * sizeof(bf16_t);            // 6.3 MB
  bf16_t* Q   = (bf16_t*)ws;
  bf16_t* K   = (bf16_t*)(ws + qBytes);
  bf16_t* Vt  = (bf16_t*)(ws + 2 * qBytes);
  bf16_t* Yb  = (bf16_t*)(ws + 2 * qBytes + vBytes);
  bf16_t* WtA = (bf16_t*)(ws + 3 * qBytes + vBytes);
  bf16_t* WtP = (bf16_t*)(ws + 3 * qBytes + vBytes + wtABytes);

  // One-time fp32->bf16 weight transposes (Wt[N][K], K-contiguous).
  transpose_w<<<dim3(48, 16), 256, 0, stream>>>(W_attn, WtA, 1024, 3072);
  transpose_w<<<dim3(16, 16), 256, 0, stream>>>(W_proj, WtP, 1024, 1024);

  qkv_gemm<<<dim3(24, 469), 128, 0, stream>>>(x, WtA, b_attn, Q, K, Vt);
  attn<<<dim3(BATCH * NH), 256, 0, stream>>>(Q, K, Vt, Yb);
  proj_gemm<<<dim3(8, 469), 128, 0, stream>>>(Yb, WtP, b_proj, out);
}
